// GNN_33809982554815
// MI455X (gfx1250) — matrix operand order repaired and checked by reading, not compiled, not run on hardware
//
#include <hip/hip_runtime.h>
#include <hip/hip_bf16.h>

#define IN_DIM  128
#define HID_DIM 128
#define OUT_DIM 64
#define GEMM_K  128   // both layers contract over 128

typedef __attribute__((ext_vector_type(16))) __bf16 v16bf;
typedef __attribute__((ext_vector_type(8)))  __bf16 v8bf;
typedef __attribute__((ext_vector_type(8)))  float  v8f;

// ---------------- utility kernels ----------------

__global__ void gcn_cvt_bf16(const float* __restrict__ s, __bf16* __restrict__ d, int n) {
    int i = blockIdx.x * blockDim.x + threadIdx.x;
    if (i < n) d[i] = (__bf16)s[i];
}

__global__ void gcn_zero_f32v4(float4* __restrict__ p, int n4) {
    int i = blockIdx.x * blockDim.x + threadIdx.x;
    if (i < n4) p[i] = make_float4(0.f, 0.f, 0.f, 0.f);
}

__global__ void gcn_init_deg(float* __restrict__ deg, int n) {
    int i = blockIdx.x * blockDim.x + threadIdx.x;
    if (i < n) deg[i] = 1.0f;   // self-loop contributes 1
}

__global__ void gcn_deg_accum(const long long* __restrict__ dst, float* __restrict__ deg, int E) {
    int e = blockIdx.x * blockDim.x + threadIdx.x;
    if (e < E) atomicAdd(&deg[(int)dst[e]], 1.0f);
}

__global__ void gcn_deg_to_dinv(float* __restrict__ d, int n) {
    int i = blockIdx.x * blockDim.x + threadIdx.x;
    if (i < n) { float v = d[i]; d[i] = (v > 0.f) ? rsqrtf(v) : 0.f; }
}

// ---------------- W packing into WMMA B-fragment order ----------------
// Wp[((nTile*(K/32) + kc)*32 + lane)*16 + j] = (bf16) W[(kc*32 + (j<8?0:16) + (lane<16?0:8) + (j&7))*Nd + nTile*16 + (lane&15)]
// so each lane's 16 B-elements for a (nTile, k-chunk) fragment are one contiguous 32B load.
__global__ void gcn_pack_w(const float* __restrict__ W, __bf16* __restrict__ Wp, int Nd) {
    int i = blockIdx.x * blockDim.x + threadIdx.x;      // total = GEMM_K * Nd
    if (i >= GEMM_K * Nd) return;
    int j     = i & 15;
    int lane  = (i >> 4) & 31;
    int kc    = (i >> 9) & 3;                           // GEMM_K/32 = 4 chunks
    int nTile = i >> 11;
    int col   = lane & 15;
    int kb    = kc * 32 + ((j >> 3) << 4) + ((lane >> 4) << 3) + (j & 7);       // lane<16 -> K 0..7,16..23, lane>=16 -> 8..15,24..31
    Wp[i] = (__bf16)W[(size_t)kb * (size_t)Nd + nTile * 16 + col];
}

// ---------------- WMMA GEMM: C[M,Nd] = A[M,128](bf16) x W[128,Nd](bf16, packed) ----------------
// One wave computes a 16x32 output strip (two 16x16 tiles sharing the A fragment).
// K fully unrolled: 4 chunks x 2 WMMAs = 8 v_wmma per wave, straight-line.

__global__ __launch_bounds__(256)
void gcn_wmma_gemm(const __bf16* __restrict__ A, const __bf16* __restrict__ Wp,
                   float* __restrict__ C, int M, int Nd) {
    const int tid     = threadIdx.x;
    const int wave    = tid >> 5;
    const int lane    = tid & 31;
    const int nPairs  = Nd >> 5;                  // 16x32 strips along N
    const int mTiles  = M >> 4;

    const int g     = blockIdx.x * 8 + wave;      // global strip id (uniform per wave)
    const int mTile = g / nPairs;
    const int nPair = g - mTile * nPairs;
    if (mTile >= mTiles) return;                  // whole wave exits; EXEC stays all-1s

    const int laneHi = lane >> 4;
    const int col    = lane & 15;
    const int nt0    = nPair * 2;
    const int nt1    = nt0 + 1;

    // A fragment (16x32 bf16): lane holds row (lane&15);
    // elems 0-7 -> K = k0 + (laneHi?8:0)+j ; elems 8-15 -> K = k0+16+(laneHi?8:0)+j (both 16B contiguous)
    const __bf16* arow = A + (size_t)(mTile * 16 + col) * (size_t)GEMM_K;

    v8f acc0 = {};
    v8f acc1 = {};
#pragma unroll
    for (int kc = 0; kc < GEMM_K / 32; ++kc) {
        const int a0 = kc * 32 + (laneHi ? 8 : 0);
        v8bf lo = *(const v8bf*)(arow + a0);
        v8bf hi = *(const v8bf*)(arow + a0 + 16);
        v16bf a = __builtin_shufflevector(lo, hi, 0,1,2,3,4,5,6,7,8,9,10,11,12,13,14,15);

        // B fragments: one contiguous 32B load per lane per tile from the packed buffer
        v16bf b0 = *(const v16bf*)(Wp + ((size_t)(nt0 * (GEMM_K / 32) + kc) * 32 + lane) * 16);
        v16bf b1 = *(const v16bf*)(Wp + ((size_t)(nt1 * (GEMM_K / 32) + kc) * 32 + lane) * 16);

        acc0 = __builtin_amdgcn_wmma_f32_16x16x32_bf16(false, a, false, b0, (short)0, acc0, false, false);
        acc1 = __builtin_amdgcn_wmma_f32_16x16x32_bf16(false, a, false, b1, (short)0, acc1, false, false);
    }

    // C/D layout: elem r -> row = r + (laneHi?8:0), col = lane&15
    float* crow = C + (size_t)(mTile * 16 + laneHi * 8) * (size_t)Nd + nt0 * 16 + col;
#pragma unroll
    for (int r = 0; r < 8; ++r) {
        crow[(size_t)r * (size_t)Nd]      = acc0[r];
        crow[(size_t)r * (size_t)Nd + 16] = acc1[r];
    }
}

// ---------------- edge scatter (segment_sum) ----------------
// Layer 1: 128 feats, one wave per edge, 4 floats/lane.
__global__ void gcn_scatter128(const long long* __restrict__ src, const long long* __restrict__ dst,
                               const float* __restrict__ dinv, const float* __restrict__ t,
                               float* __restrict__ agg, int E) {
    long long tid = (long long)blockIdx.x * blockDim.x + threadIdx.x;
    int e    = (int)(tid >> 5);
    int lane = (int)(tid & 31);
    if (e >= E) return;
    int s = (int)src[e], d = (int)dst[e];
    float w = dinv[s] * dinv[d];
    float4 v = *(const float4*)(t + (size_t)s * 128 + lane * 4);
    float* o = agg + (size_t)d * 128 + lane * 4;
    atomicAdd(o + 0, v.x * w);
    atomicAdd(o + 1, v.y * w);
    atomicAdd(o + 2, v.z * w);
    atomicAdd(o + 3, v.w * w);
}

// Layer 2: 64 feats, one wave per edge, 2 floats/lane.
__global__ void gcn_scatter64(const long long* __restrict__ src, const long long* __restrict__ dst,
                              const float* __restrict__ dinv, const float* __restrict__ t,
                              float* __restrict__ agg, int E) {
    long long tid = (long long)blockIdx.x * blockDim.x + threadIdx.x;
    int e    = (int)(tid >> 5);
    int lane = (int)(tid & 31);
    if (e >= E) return;
    int s = (int)src[e], d = (int)dst[e];
    float w = dinv[s] * dinv[d];
    float2 v = *(const float2*)(t + (size_t)s * 64 + lane * 2);
    float* o = agg + (size_t)d * 64 + lane * 2;
    atomicAdd(o + 0, v.x * w);
    atomicAdd(o + 1, v.y * w);
}

// ---------------- epilogues (self-loop + bias [+ relu]) ----------------

__global__ void gcn_epilogue1(const float* __restrict__ agg, const float* __restrict__ t1,
                              const float* __restrict__ dinv, const float* __restrict__ b1,
                              __bf16* __restrict__ hb, int total) {
    int i = blockIdx.x * blockDim.x + threadIdx.x;
    if (i >= total) return;
    int node = i >> 7, ch = i & 127;
    float di = dinv[node];
    float v = agg[i] + t1[i] * di * di + b1[ch];
    hb[i] = (__bf16)fmaxf(v, 0.0f);
}

__global__ void gcn_epilogue2(float* __restrict__ out, const float* __restrict__ t2,
                              const float* __restrict__ dinv, const float* __restrict__ b2,
                              int total) {
    int i = blockIdx.x * blockDim.x + threadIdx.x;
    if (i >= total) return;
    int node = i >> 6, ch = i & 63;
    float di = dinv[node];
    out[i] += t2[i] * di * di + b2[ch];
}

// ---------------- launch ----------------

static inline int cdiv_ll(long long a, long long b) { return (int)((a + b - 1) / b); }

extern "C" void kernel_launch(void* const* d_in, const int* in_sizes, int n_in,
                              void* d_out, int out_size, void* d_ws, size_t ws_size,
                              hipStream_t stream) {
    (void)n_in; (void)ws_size; (void)out_size;
    const float*     x    = (const float*)d_in[0];
    const long long* eidx = (const long long*)d_in[1];   // int64 (2,E)
    const float*     W1   = (const float*)d_in[2];
    const float*     b1   = (const float*)d_in[3];
    const float*     W2   = (const float*)d_in[4];
    const float*     b2   = (const float*)d_in[5];

    const int N = in_sizes[0] / IN_DIM;
    const int E = in_sizes[1] / 2;
    const long long* srcIdx = eidx;
    const long long* dstIdx = eidx + E;

    char* ws = (char*)d_ws;
    size_t off = 0;
    auto take = [&](size_t bytes) -> void* {
        void* p = ws + off;
        off = (off + bytes + 255) & ~(size_t)255;
        return p;
    };
    float*  dinv = (float*) take((size_t)N * 4);
    __bf16* xb   = (__bf16*)take((size_t)N * IN_DIM * 2);
    __bf16* w1p  = (__bf16*)take((size_t)GEMM_K * HID_DIM * 2);   // packed B-fragments
    __bf16* w2p  = (__bf16*)take((size_t)GEMM_K * OUT_DIM * 2);   // packed B-fragments
    float*  t1   = (float*) take((size_t)N * HID_DIM * 4);
    float*  agg1 = (float*) take((size_t)N * HID_DIM * 4);
    __bf16* hb   = (__bf16*)take((size_t)N * HID_DIM * 2);
    float*  t2   = (float*) take((size_t)N * OUT_DIM * 4);
    float*  out  = (float*)d_out;

    const int T = 256;

    // x -> bf16 ; W1/W2 -> packed bf16 B-fragment order
    gcn_cvt_bf16<<<cdiv_ll((long long)N * IN_DIM, T), T, 0, stream>>>(x, xb, N * IN_DIM);
    gcn_pack_w<<<cdiv_ll(GEMM_K * HID_DIM, T), T, 0, stream>>>(W1, w1p, HID_DIM);
    gcn_pack_w<<<cdiv_ll(GEMM_K * OUT_DIM, T), T, 0, stream>>>(W2, w2p, OUT_DIM);

    // degree -> dinv (in place)
    gcn_init_deg<<<cdiv_ll(N, T), T, 0, stream>>>(dinv, N);
    gcn_deg_accum<<<cdiv_ll(E, T), T, 0, stream>>>(dstIdx, dinv, E);
    gcn_deg_to_dinv<<<cdiv_ll(N, T), T, 0, stream>>>(dinv, N);

    // zero aggregation targets (atomics accumulate)
    gcn_zero_f32v4<<<cdiv_ll((long long)N * HID_DIM / 4, T), T, 0, stream>>>((float4*)agg1, N * HID_DIM / 4);
    gcn_zero_f32v4<<<cdiv_ll((long long)N * OUT_DIM / 4, T), T, 0, stream>>>((float4*)out,  N * OUT_DIM / 4);

    // layer 1: t1 = x @ W1 (WMMA), scatter, self-loop+bias+relu -> hb (bf16)
    {
        long long strips = (long long)(N / 16) * (HID_DIM / 32);
        gcn_wmma_gemm<<<cdiv_ll(strips, 8), T, 0, stream>>>(xb, w1p, t1, N, HID_DIM);
    }
    gcn_scatter128<<<cdiv_ll((long long)E * 32, T), T, 0, stream>>>(srcIdx, dstIdx, dinv, t1, agg1, E);
    gcn_epilogue1<<<cdiv_ll((long long)N * HID_DIM, T), T, 0, stream>>>(agg1, t1, dinv, b1, hb, N * HID_DIM);

    // layer 2: t2 = h @ W2 (WMMA), scatter into out, self-loop+bias
    {
        long long strips = (long long)(N / 16) * (OUT_DIM / 32);
        gcn_wmma_gemm<<<cdiv_ll(strips, 8), T, 0, stream>>>(hb, w2p, t2, N, OUT_DIM);
    }
    gcn_scatter64<<<cdiv_ll((long long)E * 32, T), T, 0, stream>>>(srcIdx, dstIdx, dinv, t2, out, E);
    gcn_epilogue2<<<cdiv_ll((long long)N * OUT_DIM, T), T, 0, stream>>>(out, t2, dinv, b2, N * OUT_DIM);
}
